// MultiheadAttention_41016937677091
// MI455X (gfx1250) — compile-verified, matrix operand order checked by reading, not run on hardware
//
#include <hip/hip_runtime.h>
#include <hip/hip_bf16.h>
#include <math.h>

// ---------------------------------------------------------------------------
// MHA forward for gfx1250 (MI455X): TDM-staged LDS + wave32 WMMA pipeline.
// ---------------------------------------------------------------------------

typedef __attribute__((ext_vector_type(16))) _Float16 v16h;
typedef __attribute__((ext_vector_type(8)))  _Float16 v8h;
typedef __attribute__((ext_vector_type(8)))  float    v8f;
typedef __attribute__((ext_vector_type(4)))  unsigned int u32x4;
typedef __attribute__((ext_vector_type(4)))  int          i32x4;
typedef __attribute__((ext_vector_type(8)))  int          i32x8;

union F16x16 { v16h v; v8h h[2]; };

#define B_  2
#define S_  2048
#define E_  1024
#define H_  16
#define DH_ 64
#define ROWS_ (B_ * S_)        // 4096 token rows
#define QKVC_ (3 * H_ * DH_)   // 3072 packed qkv output columns

// ---- blocked GEMM geometry ----
#define BM 128                 // block tile M
#define BN 128                 // block tile N
#define BK 64                  // staged K per TDM load
#define LROW 72                // LDS halves per tile row (64 + 8 pad via TDM)
#define ATILE_H (BM * LROW)    // 9216 halves per tile
#define STAGE_H (2 * ATILE_H)  // A then B
#define STAGE_B (STAGE_H * 2)  // stage bytes = 36864

// workspace offsets in _Float16 units (total 24M halves = 48MB)
#define XH_OFF    ((size_t)0)            // 4096x1024
#define WALL_OFF  ((size_t)4194304)      // 3072x1024  [Wq;Wk;Wv] (N x K)
#define PROJT_OFF ((size_t)7340032)      // 1024x1024  proj transposed (N x K)
#define Q_OFF     ((size_t)8388608)      // (B,H,S,Dh)
#define K_OFF     ((size_t)12582912)     // (B,H,S,Dh)
#define VT_OFF    ((size_t)16777216)     // (B,H,Dh,S)
#define Z_OFF     ((size_t)20971520)     // (B,H,S,Dh) == quirk-flattened (4096x1024)

__device__ __forceinline__ v8f wmma16(v16h a, v16h b, v8f c) {
  return __builtin_amdgcn_wmma_f32_16x16x32_f16(false, a, false, b, (short)0, c,
                                                false, false);
}

// Load one 16x32 f16 A/B fragment from a row-major (16 x ld) tile (global).
__device__ __forceinline__ v16h ld_frag_g(const _Float16* __restrict__ tile, int ld) {
  int lane = threadIdx.x & 31;
  const _Float16* p = tile + (size_t)(lane & 15) * ld + ((lane >> 4) << 3);
  F16x16 f;
  f.h[0] = *(const v8h*)(p);
  f.h[1] = *(const v8h*)(p + 16);
  return f.v;
}

// ---------------------------------------------------------------------------
// TDM: issue a 2D tensor_load_to_lds of a (128 rows x 64 halves) tile.
// D# per CDNA5 ISA 8.3/8.4: data_size=2B, pad 4 dwords every 32 dwords
// -> LDS row pitch 144B (LROW=72 halves). workgroup_mask=0 (no cluster).
// ---------------------------------------------------------------------------
__device__ __forceinline__ void tdm_tile_load(unsigned int lds_byte_off,
                                              const _Float16* g,
                                              unsigned int tensor_d0,
                                              unsigned int tensor_d1,
                                              unsigned long long stride0) {
  unsigned long long ga = (unsigned long long)(uintptr_t)g;
  u32x4 g0;
  g0[0] = 1u;                                   // count=1, user desc, no gather
  g0[1] = lds_byte_off;                         // lds_addr
  g0[2] = (unsigned int)ga;                     // global_addr[31:0]
  g0[3] = (unsigned int)((ga >> 32) & 0x01FFFFFFu) | (2u << 30);  // type=2

  // group1: [17:16]=data_size(1->2B) [20]=pad_en [24:22]=pad_interval(4->32dw)
  //         [31:25]=pad_amount(3->4dw) [79:48]=tensor_dim0 [111:80]=tensor_dim1
  //         [127:112]=tile_dim0(64) [143:128]=tile_dim1(128)
  //         [207:160]=tensor_dim0_stride
  unsigned long long lo = (unsigned long long)((1u << 16) | (1u << 20) |
                                               (4u << 22) | (3u << 25)) |
                          ((unsigned long long)(tensor_d0 & 0xFFFFu) << 48);
  unsigned long long hi = ((unsigned long long)tensor_d0 >> 16) |
                          ((unsigned long long)tensor_d1 << 16) |
                          ((unsigned long long)64u << 48);
  unsigned long long w2 = (unsigned long long)128u |           // tile_dim1
                          ((stride0 & 0xFFFFFFFFull) << 32);   // stride lo32
  unsigned long long w3 = (stride0 >> 32) & 0xFFFFull;         // stride hi16
  i32x8 g1;
  g1[0] = (int)(unsigned int)lo;  g1[1] = (int)(unsigned int)(lo >> 32);
  g1[2] = (int)(unsigned int)hi;  g1[3] = (int)(unsigned int)(hi >> 32);
  g1[4] = (int)(unsigned int)w2;  g1[5] = (int)(unsigned int)(w2 >> 32);
  g1[6] = (int)(unsigned int)w3;  g1[7] = (int)(unsigned int)(w3 >> 32);
  i32x4 z4 = {0, 0, 0, 0};
#if defined(__clang_major__) && (__clang_major__ >= 23)
  i32x8 z8 = {0, 0, 0, 0, 0, 0, 0, 0};
  __builtin_amdgcn_tensor_load_to_lds(g0, g1, z4, z4, z8, 0);
#else
  __builtin_amdgcn_tensor_load_to_lds(g0, g1, z4, z4, 0);
#endif
}

// ---------------------------------------------------------------------------
// Blocked GEMM core: 256 threads, block tile 128x128, wave tile 64x32.
// A,B are row-major (M x K)/(N x K); K = E_ = 1024. TDM double-buffers LDS.
// ---------------------------------------------------------------------------
__device__ __forceinline__ void gemm_core(const _Float16* __restrict__ Ag,
                                          const _Float16* __restrict__ Bg,
                                          int aRows, int bRows,
                                          int row0, int col0,
                                          _Float16* smem, v8f acc[4][2]) {
  const unsigned int lbase = (unsigned int)(uintptr_t)(void*)smem;
  const int lane  = threadIdx.x & 31;
  const int wid   = threadIdx.x >> 5;
  const int mbase = (wid >> 2) * 64;   // wave M offset in block tile
  const int nbase = (wid & 3) * 32;    // wave N offset in block tile
  const int lr    = lane & 15;
  const int kb    = (lane >> 4) << 3;
  const bool wave0 = (wid == 0);
  const int NST = E_ / BK;             // 16 stages

  if (wave0) {
    tdm_tile_load(lbase,                Ag + (size_t)row0 * E_, E_, aRows, E_);
    tdm_tile_load(lbase + ATILE_H * 2,  Bg + (size_t)col0 * E_, E_, bRows, E_);
  }

  for (int st = 0; st < NST; ++st) {
    const int cur = st & 1;
    __syncthreads();                   // all waves done reading buf(1-cur)
    if (wave0) {
      if (st + 1 < NST) {
        const int kk = (st + 1) * BK;
        const unsigned int nb = lbase + (unsigned int)(1 - cur) * STAGE_B;
        tdm_tile_load(nb,                Ag + (size_t)row0 * E_ + kk, E_, aRows, E_);
        tdm_tile_load(nb + ATILE_H * 2,  Bg + (size_t)col0 * E_ + kk, E_, bRows, E_);
        __builtin_amdgcn_s_wait_tensorcnt((short)2);   // stage st's pair done
      } else {
        __builtin_amdgcn_s_wait_tensorcnt((short)0);
      }
    }
    __syncthreads();                   // buf(cur) visible to all waves

    const _Float16* As = smem + (size_t)cur * STAGE_H;
    const _Float16* Bs = As + ATILE_H;
#pragma unroll
    for (int ks = 0; ks < BK; ks += 32) {
      v16h af[4], bf[2];
#pragma unroll
      for (int mi = 0; mi < 4; ++mi) {
        const _Float16* p = As + (size_t)(mbase + mi * 16 + lr) * LROW + ks + kb;
        F16x16 f; f.h[0] = *(const v8h*)p; f.h[1] = *(const v8h*)(p + 16);
        af[mi] = f.v;
      }
#pragma unroll
      for (int ni = 0; ni < 2; ++ni) {
        const _Float16* p = Bs + (size_t)(nbase + ni * 16 + lr) * LROW + ks + kb;
        F16x16 f; f.h[0] = *(const v8h*)p; f.h[1] = *(const v8h*)(p + 16);
        bf[ni] = f.v;
      }
#pragma unroll
      for (int mi = 0; mi < 4; ++mi)
#pragma unroll
        for (int ni = 0; ni < 2; ++ni)
          acc[mi][ni] = wmma16(af[mi], bf[ni], acc[mi][ni]);
    }
  }
}

// ---------------- conversion / packing kernels ----------------
__global__ void cvt_f32_to_f16(const float* __restrict__ src,
                               _Float16* __restrict__ dst, int n) {
  int i = blockIdx.x * blockDim.x + threadIdx.x;
  if (i < n) dst[i] = (_Float16)src[i];
}

__global__ void transpose_f32_to_f16(const float* __restrict__ src,  // (K x N)
                                     _Float16* __restrict__ dst,     // (N x K)
                                     int K, int N) {
  int i = blockIdx.x * blockDim.x + threadIdx.x;
  if (i < K * N) {
    int k = i / N, n = i % N;
    dst[(size_t)n * K + k] = (_Float16)src[i];
  }
}

// ---------------- fused QKV projection: (4096x3072) ----------
__global__ __launch_bounds__(256)
void qkv_gemm(const _Float16* __restrict__ Xh,    // 4096x1024
              const _Float16* __restrict__ Wall,  // 3072x1024 (N x K)
              const float* __restrict__ bq, const float* __restrict__ bk,
              const float* __restrict__ bv,
              _Float16* __restrict__ Qb,  // (B,H,S,Dh)
              _Float16* __restrict__ Kb,  // (B,H,S,Dh)
              _Float16* __restrict__ Vt)  // (B,H,Dh,S)
{
  __shared__ _Float16 smem[2 * STAGE_H];
  const int row0 = blockIdx.y * BM;
  const int col0 = blockIdx.x * BN;

  v8f acc[4][2] = {};
  gemm_core(Xh, Wall, ROWS_, QKVC_, row0, col0, smem, acc);

  const int lane  = threadIdx.x & 31;
  const int wid   = threadIdx.x >> 5;
  const int mbase = (wid >> 2) * 64;
  const int nbase = (wid & 3) * 32;
  const int ncol  = lane & 15;
  const int hoff  = (lane >> 4) << 3;

#pragma unroll
  for (int ni = 0; ni < 2; ++ni) {
    const int col = col0 + nbase + ni * 16 + ncol;   // 0..3071
    const int mt  = col >> 10;                       // 0=q 1=k 2=v
    const int idx = col & 1023;                      // h*64+d
    const int h   = idx >> 6;
    const int d   = idx & 63;
    const float bias = (mt == 0 ? bq : mt == 1 ? bk : bv)[idx];
#pragma unroll
    for (int mi = 0; mi < 4; ++mi) {
#pragma unroll
      for (int r = 0; r < 8; ++r) {
        const int row = row0 + mbase + mi * 16 + hoff + r;   // 0..4095
        const int b_  = row >> 11;
        const int s   = row & 2047;
        const _Float16 val = (_Float16)(acc[mi][ni][r] + bias);
        const size_t bh = (size_t)(b_ * H_ + h);
        if (mt == 0)      Qb[(bh * S_ + s) * DH_ + d] = val;
        else if (mt == 1) Kb[(bh * S_ + s) * DH_ + d] = val;
        else              Vt[(bh * DH_ + d) * S_ + s] = val;
      }
    }
  }
}

// ---------------- output projection: (4096x1024) -> f32 ------------
__global__ __launch_bounds__(256)
void proj_gemm(const _Float16* __restrict__ Zb,     // 4096x1024 (quirk-flat)
               const _Float16* __restrict__ projT,  // 1024x1024 (N x K)
               float* __restrict__ out)             // 4096x1024 f32
{
  __shared__ _Float16 smem[2 * STAGE_H];
  const int row0 = blockIdx.y * BM;
  const int col0 = blockIdx.x * BN;

  v8f acc[4][2] = {};
  gemm_core(Zb, projT, ROWS_, E_, row0, col0, smem, acc);

  const int lane  = threadIdx.x & 31;
  const int wid   = threadIdx.x >> 5;
  const int mbase = (wid >> 2) * 64;
  const int nbase = (wid & 3) * 32;
  const int ncol  = lane & 15;
  const int hoff  = (lane >> 4) << 3;

#pragma unroll
  for (int ni = 0; ni < 2; ++ni) {
    const int col = col0 + nbase + ni * 16 + ncol;
#pragma unroll
    for (int mi = 0; mi < 4; ++mi)
#pragma unroll
      for (int r = 0; r < 8; ++r)
        out[(size_t)(row0 + mbase + mi * 16 + hoff + r) * E_ + col] = acc[mi][ni][r];
  }
}

// ---------------- flash attention: one wave per (b,h, 16-row strip) ----------
__global__ __launch_bounds__(32)
void attn_kernel(const _Float16* __restrict__ Qb,
                 const _Float16* __restrict__ Kb,
                 const _Float16* __restrict__ Vt,
                 _Float16* __restrict__ Zb)
{
  __shared__ _Float16 attLds[16 * 32];   // att tile bounce C-layout -> A-layout

  const int lane = threadIdx.x;
  const int bh   = blockIdx.y;           // b*H + h
  const int s0   = blockIdx.x * 16;      // query rows [s0, s0+16)
  const int ncol = lane & 15;
  const int hoff = (lane >> 4) << 3;

  const _Float16* Qh = Qb + (size_t)bh * S_ * DH_;
  const _Float16* Kh = Kb + (size_t)bh * S_ * DH_;
  const _Float16* Vh = Vt + (size_t)bh * DH_ * S_;

  const v16h qf0 = ld_frag_g(Qh + (size_t)s0 * DH_ + 0,  DH_);
  const v16h qf1 = ld_frag_g(Qh + (size_t)s0 * DH_ + 32, DH_);

  v8f acc[4] = {v8f{}, v8f{}, v8f{}, v8f{}};
  float m[8], l[8];
#pragma unroll
  for (int r = 0; r < 8; ++r) { m[r] = -__builtin_inff(); l[r] = 0.f; }

  for (int kt = 0; kt < s0 + 16; kt += 32) {   // 32 keys per chunk
    v16h k0a = ld_frag_g(Kh + (size_t)kt * DH_ + 0,  DH_);
    v16h k0b = ld_frag_g(Kh + (size_t)kt * DH_ + 32, DH_);
    v16h k1a = ld_frag_g(Kh + (size_t)(kt + 16) * DH_ + 0,  DH_);
    v16h k1b = ld_frag_g(Kh + (size_t)(kt + 16) * DH_ + 32, DH_);
    v8f st0 = {}; st0 = wmma16(qf0, k0a, st0); st0 = wmma16(qf1, k0b, st0);
    v8f st1 = {}; st1 = wmma16(qf0, k1a, st1); st1 = wmma16(qf1, k1b, st1);

    float p0[8], p1[8], alpha[8];
#pragma unroll
    for (int r = 0; r < 8; ++r) {
      const int row = s0 + hoff + r;
      float a0 = st0[r] * 0.125f;
      float a1 = st1[r] * 0.125f;
      if (kt + ncol      > row) a0 = -__builtin_inff();
      if (kt + 16 + ncol > row) a1 = -__builtin_inff();
      float cm = fmaxf(a0, a1);
#pragma unroll
      for (int sh = 1; sh < 16; sh <<= 1) cm = fmaxf(cm, __shfl_xor(cm, sh, 32));
      const float nm = fmaxf(m[r], cm);
      alpha[r] = __expf(m[r] - nm);
      m[r] = nm;
      p0[r] = __expf(a0 - nm);
      p1[r] = __expf(a1 - nm);
      float s = p0[r] + p1[r];
#pragma unroll
      for (int sh = 1; sh < 16; sh <<= 1) s += __shfl_xor(s, sh, 32);
      l[r] = l[r] * alpha[r] + s;
    }
#pragma unroll
    for (int nb = 0; nb < 4; ++nb)
#pragma unroll
      for (int r = 0; r < 8; ++r) acc[nb][r] *= alpha[r];

#pragma unroll
    for (int r = 0; r < 8; ++r) {
      attLds[(hoff + r) * 32 + ncol]      = (_Float16)p0[r];
      attLds[(hoff + r) * 32 + 16 + ncol] = (_Float16)p1[r];
    }
    __syncthreads();
    const _Float16* lp = &attLds[(size_t)(lane & 15) * 32 + ((lane >> 4) << 3)];
    F16x16 af;
    af.h[0] = *(const v8h*)(lp);
    af.h[1] = *(const v8h*)(lp + 16);
    __syncthreads();

#pragma unroll
    for (int nb = 0; nb < 4; ++nb) {
      v16h vf = ld_frag_g(Vh + (size_t)(nb * 16) * S_ + kt, S_);
      acc[nb] = wmma16(af.v, vf, acc[nb]);
    }
  }

  _Float16* Zh = Zb + (size_t)bh * S_ * DH_;
#pragma unroll
  for (int r = 0; r < 8; ++r) {
    const float inv = 1.f / (l[r] * 8.f);
    const int row = s0 + hoff + r;
#pragma unroll
    for (int nb = 0; nb < 4; ++nb)
      Zh[(size_t)row * DH_ + nb * 16 + ncol] = (_Float16)(acc[nb][r] * inv);
  }
}

// ---------------------------------------------------------------------------
extern "C" void kernel_launch(void* const* d_in, const int* in_sizes, int n_in,
                              void* d_out, int out_size, void* d_ws, size_t ws_size,
                              hipStream_t stream) {
  const float* x    = (const float*)d_in[0];
  const float* Wq   = (const float*)d_in[1];
  const float* bq   = (const float*)d_in[2];
  const float* Wk   = (const float*)d_in[3];
  const float* bk   = (const float*)d_in[4];
  const float* Wv   = (const float*)d_in[5];
  const float* bv   = (const float*)d_in[6];
  const float* proj = (const float*)d_in[7];
  float* out        = (float*)d_out;

  _Float16* ws = (_Float16*)d_ws;
  _Float16* Xh    = ws + XH_OFF;
  _Float16* Wall  = ws + WALL_OFF;
  _Float16* projT = ws + PROJT_OFF;
  _Float16* Qb    = ws + Q_OFF;
  _Float16* Kb    = ws + K_OFF;
  _Float16* Vt    = ws + VT_OFF;
  _Float16* Zb    = ws + Z_OFF;

  const int nX = ROWS_ * E_;          // 4,194,304
  const int nW = H_ * DH_ * E_;       // 1,048,576 per weight

  cvt_f32_to_f16<<<(nX + 255) / 256, 256, 0, stream>>>(x, Xh, nX);
  cvt_f32_to_f16<<<(nW + 255) / 256, 256, 0, stream>>>(Wq, Wall + (size_t)0 * nW, nW);
  cvt_f32_to_f16<<<(nW + 255) / 256, 256, 0, stream>>>(Wk, Wall + (size_t)1 * nW, nW);
  cvt_f32_to_f16<<<(nW + 255) / 256, 256, 0, stream>>>(Wv, Wall + (size_t)2 * nW, nW);
  transpose_f32_to_f16<<<(E_ * E_ + 255) / 256, 256, 0, stream>>>(proj, projT, E_, E_);

  // QKV: 32 row-blocks x 24 col-blocks, 256 threads (8 waves) each
  qkv_gemm<<<dim3(QKVC_ / BN, ROWS_ / BM), 256, 0, stream>>>(
      Xh, Wall, bq, bk, bv, Qb, Kb, Vt);

  // attention: (S/16) strips x (B*H) heads
  attn_kernel<<<dim3(S_ / 16, B_ * H_), 32, 0, stream>>>(Qb, Kb, Vt, Zb);

  // projection: 32 row-blocks x 8 col-blocks
  proj_gemm<<<dim3(E_ / BN, ROWS_ / BM), 256, 0, stream>>>(Zb, projT, out);
}
